// MultiHeadSelfAttention_858993459806
// MI455X (gfx1250) — matrix operand order repaired and checked by reading, not compiled, not run on hardware
//
#include <hip/hip_runtime.h>
#include <stdint.h>

#define B_  8
#define S_  1024
#define E_  768
#define H_  12
#define D_  64

typedef __bf16 bf16_t;
typedef bf16_t v16bf __attribute__((ext_vector_type(16)));
typedef float  v8f   __attribute__((ext_vector_type(8)));

typedef unsigned int u32x4 __attribute__((ext_vector_type(4)));
typedef int          i32x4 __attribute__((ext_vector_type(4)));
typedef int          i32x8 __attribute__((ext_vector_type(8)));

union ABu { v16bf v; unsigned u[8]; };

#if __has_builtin(__builtin_amdgcn_tensor_load_to_lds) && __has_builtin(__builtin_amdgcn_s_wait_tensorcnt)
#define USE_TDM 1
#else
#define USE_TDM 0
#endif

static __device__ __forceinline__ unsigned short f32_bf16(float f) {
  union { float f; unsigned u; } c; c.f = f;
  unsigned u = c.u + 0x7FFFu + ((c.u >> 16) & 1u);   // round-to-nearest-even
  return (unsigned short)(u >> 16);
}

// 16-bit A-matrix 16x32 layout: lanes 0-15 hold K={2j,2j+1} (j<4) / {16+2(j-4),..} (j>=4);
// lanes 16-31 same +8.
static __device__ __forceinline__ int a_koff(int j, int hi) {
  return ((j < 4) ? (2 * j) : (16 + 2 * (j - 4))) + (hi ? 8 : 0);
}

#if USE_TDM
// Issue a 2D TDM load: tile tileK x tileN (elements, 2-byte data) from a row-major
// [tdim1 x tdim0] bf16 tensor with row stride stride0 (elements) into LDS at ldsOff.
// D# packing per CDNA5 ISA ch.8.3/8.4 (group0: count/lds/global/type, group1: dims).
static __device__ __forceinline__ void tdm_load_2d(const void* gaddr, unsigned ldsOff,
                                                   unsigned tileK, unsigned tileN,
                                                   unsigned tdim0, unsigned tdim1,
                                                   unsigned stride0) {
  const uint64_t ga = (uint64_t)(uintptr_t)gaddr;
  union { unsigned u[4]; u32x4 v; } g0;
  g0.u[0] = 1u;                                            // count=1, user mode
  g0.u[1] = ldsOff;                                        // lds_addr (bytes)
  g0.u[2] = (unsigned)ga;                                  // global_addr[31:0]
  g0.u[3] = (unsigned)((ga >> 32) & 0x01FFFFFFu) | (2u << 30);  // addr[56:32] | type=2
  union { int i[8]; i32x8 v; } g1;
  g1.i[0] = (int)(1u << 16);                               // data_size=1 (2 bytes)
  g1.i[1] = (int)((tdim0 & 0xFFFFu) << 16);                // tensor_dim0[15:0]
  g1.i[2] = (int)(((tdim0 >> 16) & 0xFFFFu) | ((tdim1 & 0xFFFFu) << 16));
  g1.i[3] = (int)(((tdim1 >> 16) & 0xFFFFu) | ((tileK & 0xFFFFu) << 16)); // tile_dim0
  g1.i[4] = (int)(tileN & 0xFFFFu);                        // tile_dim1 (tile_dim2=0)
  g1.i[5] = (int)stride0;                                  // tensor_dim0_stride[31:0]
  g1.i[6] = 0;
  g1.i[7] = 0;
  i32x4 z4 = {0, 0, 0, 0};
#if __clang_major__ >= 23
  i32x8 z8 = {0, 0, 0, 0, 0, 0, 0, 0};
  __builtin_amdgcn_tensor_load_to_lds(g0.v, g1.v, z4, z4, z8, 0);
#else
  __builtin_amdgcn_tensor_load_to_lds(g0.v, g1.v, z4, z4, 0);
#endif
}
#endif

// ---------------------------------------------------------------------------
// Converts
// ---------------------------------------------------------------------------
__global__ void cvt_f32_bf16(const float* __restrict__ x,
                             unsigned short* __restrict__ y, int n) {
  int i = blockIdx.x * blockDim.x + threadIdx.x;
  if (i < n) y[i] = f32_bf16(x[i]);
}

// wt[c*E + k] = w[k*E + c]  (weights stored [N,K] so B-frag pairs are contiguous)
__global__ void cvt_w_transpose(const float* __restrict__ w,
                                unsigned short* __restrict__ wt) {
  int o = blockIdx.x * blockDim.x + threadIdx.x;
  if (o >= E_ * E_) return;
  int c = o / E_, k = o - c * E_;
  wt[o] = f32_bf16(w[k * E_ + c]);
}

// ---------------------------------------------------------------------------
// bf16 WMMA GEMM: C[M,N] = A[M,K] @ Bt[N,K]^T + bias
// B tile (64 cols x 32 k) staged in LDS by the Tensor Data Mover, double
// buffered; A fragments software-pipelined in registers.
// Wave computes 32x64 (2 A frags x 4 accumulators = 8 back-to-back WMMAs per
// k-step, each B frag reused twice); block = 8 waves -> 256x64 tile.
// mode 0: store bf16 row-major; mode 1: bf16 into Vt[B,H,D,S]; mode 2: f32.
// ---------------------------------------------------------------------------
__global__ __launch_bounds__(256)
void gemm_bf16_wmma(const unsigned short* __restrict__ A,
                    const unsigned short* __restrict__ Bt,
                    const float* __restrict__ bias,
                    unsigned short* __restrict__ outB,
                    float* __restrict__ outF,
                    int M, int N, int K, int mode) {
  __shared__ unsigned short btile[2][64 * 32];   // [buf][n_local*32 + k_local]

  const int wave   = threadIdx.x >> 5;
  const int lane   = threadIdx.x & 31;
  const int lane16 = lane & 15;
  const int hi     = lane >> 4;
  const int rowBase = blockIdx.x * 256 + wave * 32;
  const int colBase = blockIdx.y * 64;

  v8f acc[2][4];
#pragma unroll
  for (int g = 0; g < 2; ++g)
#pragma unroll
    for (int f = 0; f < 4; ++f)
#pragma unroll
      for (int r = 0; r < 8; ++r) acc[g][f][r] = 0.0f;

  const unsigned short* arow0 = A + (size_t)(rowBase + lane16) * K;
  const unsigned short* arow1 = A + (size_t)(rowBase + 16 + lane16) * K;

#if USE_TDM
  const unsigned ldsOff[2] = { (unsigned)(uintptr_t)(void*)&btile[0][0],
                               (unsigned)(uintptr_t)(void*)&btile[1][0] };
  if (wave == 0)  // preload tile 0
    tdm_load_2d(Bt + (size_t)colBase * K, ldsOff[0], 32, 64, (unsigned)K, (unsigned)N,
                (unsigned)K);
#endif

  // prefetch A fragments for kb = 0
  ABu a_cur[2], a_nxt[2];
#pragma unroll
  for (int j = 0; j < 8; ++j) {
    const int ko = a_koff(j, hi);
    a_cur[0].u[j] = *(const unsigned*)(arow0 + ko);
    a_cur[1].u[j] = *(const unsigned*)(arow1 + ko);
  }
  a_nxt[0] = a_cur[0]; a_nxt[1] = a_cur[1];

  int buf = 0;
  for (int kb = 0; kb < K; kb += 32, buf ^= 1) {
#if USE_TDM
    if (wave == 0) {
      if (kb + 32 < K) {
        tdm_load_2d(Bt + (size_t)colBase * K + kb + 32, ldsOff[buf ^ 1],
                    32, 64, (unsigned)K, (unsigned)N, (unsigned)K);
        __builtin_amdgcn_s_wait_tensorcnt(1);   // current tile landed
      } else {
        __builtin_amdgcn_s_wait_tensorcnt(0);
      }
    }
    __syncthreads();                            // btile[buf] ready for all waves
#else
    __syncthreads();
    for (int i = threadIdx.x; i < 64 * 16; i += 256) {  // cooperative copy (u32 pairs)
      const int rr = i >> 4, kk = i & 15;
      ((unsigned*)&btile[buf][0])[rr * 16 + kk] =
          *(const unsigned*)(Bt + (size_t)(colBase + rr) * K + kb + 2 * kk);
    }
    __syncthreads();
#endif

    // issue A loads for the next k-step before consuming this one
    if (kb + 32 < K) {
#pragma unroll
      for (int j = 0; j < 8; ++j) {
        const int ko = kb + 32 + a_koff(j, hi);
        a_nxt[0].u[j] = *(const unsigned*)(arow0 + ko);
        a_nxt[1].u[j] = *(const unsigned*)(arow1 + ko);
      }
    }

    // load all four B fragments first, then 8 back-to-back WMMAs
    ABu bb[4];
    const unsigned short* bl = &btile[buf][0];
#pragma unroll
    for (int f = 0; f < 4; ++f) {
      const unsigned short* brow = bl + (f * 16 + lane16) * 32;
#pragma unroll
      for (int j = 0; j < 8; ++j)
        bb[f].u[j] = *(const unsigned*)(brow + a_koff(j, hi));
    }
#pragma unroll
    for (int f = 0; f < 4; ++f) {
      acc[0][f] = __builtin_amdgcn_wmma_f32_16x16x32_bf16(
          false, a_cur[0].v, false, bb[f].v, (short)0, acc[0][f], false, false);
      acc[1][f] = __builtin_amdgcn_wmma_f32_16x16x32_bf16(
          false, a_cur[1].v, false, bb[f].v, (short)0, acc[1][f], false, false);
    }
    a_cur[0] = a_nxt[0]; a_cur[1] = a_nxt[1];
    __syncthreads();                            // all waves done with btile[buf]
  }

#pragma unroll
  for (int g = 0; g < 2; ++g)
#pragma unroll
    for (int f = 0; f < 4; ++f) {
      const int col = colBase + f * 16 + lane16;
      const float badd = bias ? bias[col] : 0.0f;
#pragma unroll
      for (int r = 0; r < 8; ++r) {
        const int row = rowBase + g * 16 + r + (hi ? 8 : 0);
        const float val = acc[g][f][r] + badd;
        if (mode == 0) {
          outB[(size_t)row * N + col] = f32_bf16(val);
        } else if (mode == 1) {
          const int bb_ = row >> 10, ss = row & 1023;
          const int hh  = col >> 6,  dd = col & 63;
          outB[(((size_t)bb_ * H_ + hh) * D_ + dd) * S_ + ss] = f32_bf16(val);
        } else {
          outF[(size_t)row * N + col] = val;
        }
      }
    }
}

// ---------------------------------------------------------------------------
// Flash attention, one wave per (b, h, 16-query tile). S=1024, D=64.
// Q,K: bf16 [B*S, E] (head slice at h*64). Vt: bf16 [B,H,D,S]. Out: bf16 [B*S, E].
// K fragments are software-pipelined across key blocks (ping-pong registers,
// loop unrolled 2x so there are no register copies); V fragments are loaded
// before the softmax so global latency overlaps the exp/shuffle work.
// ---------------------------------------------------------------------------
__global__ __launch_bounds__(256)
void flash_attn_wmma(const unsigned short* __restrict__ Q,
                     const unsigned short* __restrict__ Km,
                     const unsigned short* __restrict__ Vt,
                     unsigned short* __restrict__ Oc) {
  __shared__ unsigned short pstage[8][16 * 32];   // per-wave P tile (row-major 16x32)

  const int wave   = threadIdx.x >> 5;
  const int lane   = threadIdx.x & 31;
  const int lane16 = lane & 15;
  const int hi     = lane >> 4;

  const int task  = blockIdx.x * 8 + wave;        // 768 blocks * 8 waves = 6144 tasks
  const int qt    = task & 63;
  const int h     = (task >> 6) % H_;
  const int b     = task / (64 * H_);
  const int qbase = qt * 16;

  // Q A-fragments: 16x64 = two 16x32 fragments
  ABu aq0, aq1;
  {
    const unsigned short* qrow =
        Q + (size_t)(b * S_ + qbase + lane16) * E_ + h * D_;
#pragma unroll
    for (int j = 0; j < 8; ++j) {
      const int ko = a_koff(j, hi);
      aq0.u[j] = *(const unsigned*)(qrow + ko);
      aq1.u[j] = *(const unsigned*)(qrow + 32 + ko);
    }
  }

  float m_r[8], l_r[8];
  v8f o[4];
#pragma unroll
  for (int r = 0; r < 8; ++r) { m_r[r] = -3.0e38f; l_r[r] = 0.0f; }
#pragma unroll
  for (int f = 0; f < 4; ++f)
#pragma unroll
    for (int r = 0; r < 8; ++r) o[f][r] = 0.0f;

  unsigned short* pst = pstage[wave];

  // K B-fragments for a 32-key block: [h0 d0..31, h0 d32..63, h1 d0..31, h1 d32..63]
  auto load_kfrags = [&](int kb, ABu* kf) {
    const unsigned short* krow0 =
        Km + (size_t)(b * S_ + kb + lane16) * E_ + h * D_;
    const unsigned short* krow1 = krow0 + (size_t)16 * E_;
#pragma unroll
    for (int j = 0; j < 8; ++j) {
      kf[0].u[j] = *(const unsigned*)(krow0 + a_koff(j, hi));
      kf[1].u[j] = *(const unsigned*)(krow0 + 32 + a_koff(j, hi));
      kf[2].u[j] = *(const unsigned*)(krow1 + a_koff(j, hi));
      kf[3].u[j] = *(const unsigned*)(krow1 + 32 + a_koff(j, hi));
    }
  };

  // one 32-key flash step: consume kc, prefetch kb+32 into kn
  auto step = [&](int kb, ABu* kc, ABu* kn) {
    if (kb + 32 < S_) load_kfrags(kb + 32, kn);   // in flight during this step

    v8f s0, s1;
#pragma unroll
    for (int r = 0; r < 8; ++r) { s0[r] = 0.0f; s1[r] = 0.0f; }
    s0 = __builtin_amdgcn_wmma_f32_16x16x32_bf16(false, aq0.v, false, kc[0].v, (short)0, s0, false, false);
    s0 = __builtin_amdgcn_wmma_f32_16x16x32_bf16(false, aq1.v, false, kc[1].v, (short)0, s0, false, false);
    s1 = __builtin_amdgcn_wmma_f32_16x16x32_bf16(false, aq0.v, false, kc[2].v, (short)0, s1, false, false);
    s1 = __builtin_amdgcn_wmma_f32_16x16x32_bf16(false, aq1.v, false, kc[3].v, (short)0, s1, false, false);

    // prefetch V fragments; consumed after the softmax
    ABu bv[4];
#pragma unroll
    for (int f = 0; f < 4; ++f) {
      const unsigned short* vrow =
          Vt + (((size_t)b * H_ + h) * D_ + f * 16 + lane16) * S_ + kb;
#pragma unroll
      for (int j = 0; j < 8; ++j)
        bv[f].u[j] = *(const unsigned*)(vrow + a_koff(j, hi));
    }

    // online softmax update (per row; rows r and r+8 across lane halves)
#pragma unroll
    for (int r = 0; r < 8; ++r) {
      const float x0 = s0[r] * 0.125f;   // 1/sqrt(64)
      const float x1 = s1[r] * 0.125f;
      float mx = fmaxf(x0, x1);
      mx = fmaxf(mx, __shfl_xor(mx, 1, 32));
      mx = fmaxf(mx, __shfl_xor(mx, 2, 32));
      mx = fmaxf(mx, __shfl_xor(mx, 4, 32));
      mx = fmaxf(mx, __shfl_xor(mx, 8, 32));
      const float mnew  = fmaxf(m_r[r], mx);
      const float alpha = __expf(m_r[r] - mnew);
      m_r[r] = mnew;
      const float p0 = __expf(x0 - mnew);
      const float p1 = __expf(x1 - mnew);
      float rs = p0 + p1;
      rs += __shfl_xor(rs, 1, 32);
      rs += __shfl_xor(rs, 2, 32);
      rs += __shfl_xor(rs, 4, 32);
      rs += __shfl_xor(rs, 8, 32);
      l_r[r] = l_r[r] * alpha + rs;
      o[0][r] *= alpha; o[1][r] *= alpha; o[2][r] *= alpha; o[3][r] *= alpha;
      const int prow = r + (hi ? 8 : 0);
      pst[prow * 32 + lane16]      = f32_bf16(p0);
      pst[prow * 32 + 16 + lane16] = f32_bf16(p1);
    }

    // wave-local LDS RAW fence (DS ops from same wave are in order; be explicit)
    asm volatile("s_wait_dscnt 0" ::: "memory");

    // reload P in A-fragment layout
    ABu ap;
#pragma unroll
    for (int j = 0; j < 8; ++j)
      ap.u[j] = *(const unsigned*)(pst + lane16 * 32 + a_koff(j, hi));

    // O(16x64) += P(16x32) @ V(32x64)
#pragma unroll
    for (int f = 0; f < 4; ++f)
      o[f] = __builtin_amdgcn_wmma_f32_16x16x32_bf16(
          false, ap.v, false, bv[f].v, (short)0, o[f], false, false);
  };

  ABu ka[4], kb2[4];
  load_kfrags(0, ka);
  for (int kb = 0; kb < S_; kb += 64) {   // 2x unrolled ping-pong, no reg copies
    step(kb, ka, kb2);
    step(kb + 32, kb2, ka);
  }

  // epilogue: divide by row sums, store concat-layout bf16 [B,S,H*D]
#pragma unroll
  for (int r = 0; r < 8; ++r) {
    const float inv = 1.0f / l_r[r];
    const int row = qbase + r + (hi ? 8 : 0);
    unsigned short* orow = Oc + (size_t)(b * S_ + row) * E_ + h * D_ + lane16;
    orow[0]  = f32_bf16(o[0][r] * inv);
    orow[16] = f32_bf16(o[1][r] * inv);
    orow[32] = f32_bf16(o[2][r] * inv);
    orow[48] = f32_bf16(o[3][r] * inv);
  }
}

// ---------------------------------------------------------------------------
// Host launch
// ---------------------------------------------------------------------------
extern "C" void kernel_launch(void* const* d_in, const int* in_sizes, int n_in,
                              void* d_out, int out_size, void* d_ws, size_t ws_size,
                              hipStream_t stream) {
  (void)in_sizes; (void)n_in; (void)out_size; (void)ws_size;

  const float* X  = (const float*)d_in[0];
  const float* Wq = (const float*)d_in[1];
  const float* bq = (const float*)d_in[2];
  const float* Wk = (const float*)d_in[3];
  const float* bk = (const float*)d_in[4];
  const float* Wv = (const float*)d_in[5];
  const float* bv = (const float*)d_in[6];
  const float* Wo = (const float*)d_in[7];
  const float* bo = (const float*)d_in[8];

  const size_t XN = (size_t)B_ * S_ * E_;   // 6,291,456
  const size_t WN = (size_t)E_ * E_;        // 589,824

  uint8_t* ws = (uint8_t*)d_ws;
  unsigned short* Xb  = (unsigned short*)ws; ws += XN * 2;  // X bf16; reused as attn concat
  unsigned short* Wqb = (unsigned short*)ws; ws += WN * 2;
  unsigned short* Wkb = (unsigned short*)ws; ws += WN * 2;
  unsigned short* Wvb = (unsigned short*)ws; ws += WN * 2;
  unsigned short* Wob = (unsigned short*)ws; ws += WN * 2;
  unsigned short* Qb  = (unsigned short*)ws; ws += XN * 2;
  unsigned short* Kb  = (unsigned short*)ws; ws += XN * 2;
  unsigned short* Vt  = (unsigned short*)ws; ws += XN * 2;  // [B,H,D,S]

  cvt_f32_bf16<<<(unsigned)((XN + 255) / 256), 256, 0, stream>>>(X, Xb, (int)XN);
  cvt_w_transpose<<<(unsigned)((WN + 255) / 256), 256, 0, stream>>>(Wq, Wqb);
  cvt_w_transpose<<<(unsigned)((WN + 255) / 256), 256, 0, stream>>>(Wk, Wkb);
  cvt_w_transpose<<<(unsigned)((WN + 255) / 256), 256, 0, stream>>>(Wv, Wvb);
  cvt_w_transpose<<<(unsigned)((WN + 255) / 256), 256, 0, stream>>>(Wo, Wob);

  dim3 ggrid(8192 / 256, 768 / 64);   // 32 x 12 blocks of 256 threads
  gemm_bf16_wmma<<<ggrid, 256, 0, stream>>>(Xb, Wqb, bq, Qb, nullptr, 8192, 768, 768, 0);
  gemm_bf16_wmma<<<ggrid, 256, 0, stream>>>(Xb, Wkb, bk, Kb, nullptr, 8192, 768, 768, 0);
  gemm_bf16_wmma<<<ggrid, 256, 0, stream>>>(Xb, Wvb, bv, Vt, nullptr, 8192, 768, 768, 1);

  flash_attn_wmma<<<6144 / 8, 256, 0, stream>>>(Qb, Kb, Vt, Xb);  // writes attn into Xb

  gemm_bf16_wmma<<<ggrid, 256, 0, stream>>>(Xb, Wob, bo, nullptr, (float*)d_out,
                                            8192, 768, 768, 2);
}
